// ReferenceDecoderLayer_59502476918793
// MI455X (gfx1250) — matrix operand order repaired and checked by reading, not compiled, not run on hardware
//
#include <hip/hip_runtime.h>
#include <hip/hip_bf16.h>

// ---------------- problem constants ----------------
constexpr int Bb   = 2;
constexpr int Ss   = 1024;
constexpr int Hh   = 1024;
constexpr int NHq  = 16;
constexpr int NKVv = 4;
constexpr int HDd  = 64;
constexpr int Ee   = 8;
constexpr int Ii   = 1024;
constexpr int TDIM = Bb * Ss;          // 2048 tokens
constexpr int QKVN = NHq*HDd + 2*NKVv*HDd;   // 1536
constexpr float ALPHAc = 1.702f;
constexpr float LIMITc = 7.0f;
constexpr float NEGc   = -1.0e9f;
constexpr float SCALEc = 0.125f;       // 64^-0.5
constexpr float EPSc   = 1e-6f;

typedef float  v8f    __attribute__((ext_vector_type(8)));
typedef __bf16 bf16x8 __attribute__((ext_vector_type(8)));
typedef __bf16 bf16x16 __attribute__((ext_vector_type(16)));
typedef __bf16 bf16_t;

union B16U { bf16x16 v; bf16x8 h[2]; };

__device__ inline v8f zero8() {
  v8f c;
#pragma unroll
  for (int i = 0; i < 8; ++i) c[i] = 0.f;
  return c;
}

__device__ inline v8f wmma_bf(bf16x16 a, bf16x16 b, v8f c) {
  return __builtin_amdgcn_wmma_f32_16x16x32_bf16(
      /*neg_a=*/false, a, /*neg_b=*/false, b,
      /*c_mod=*/(short)0, c, /*reuse_a=*/false, /*reuse_b=*/false);
}

// A-operand (16x32 bf16): lane holds row, K chunks {kk+8*grp .. +8} and {kk+16+8*grp .. +8}
__device__ inline bf16x16 load_a32(const bf16_t* __restrict__ rowp, int kk, int grp) {
  B16U u;
  u.h[0] = *reinterpret_cast<const bf16x8*>(rowp + kk + 8*grp);
  u.h[1] = *reinterpret_cast<const bf16x8*>(rowp + kk + 16 + 8*grp);
  return u.v;
}
// B-operand (32x16 bf16): lane holds column, K chunks {kk+8*grp .. +8} and {kk+16+8*grp .. +8}
__device__ inline bf16x16 load_b32(const bf16_t* __restrict__ colp, int kk, int grp) {
  B16U u;
  u.h[0] = *reinterpret_cast<const bf16x8*>(colp + kk + 8*grp);
  u.h[1] = *reinterpret_cast<const bf16x8*>(colp + kk + 16 + 8*grp);
  return u.v;
}

__device__ inline float rowmax16(float v) {
#pragma unroll
  for (int o = 1; o < 16; o <<= 1) v = fmaxf(v, __shfl_xor(v, o, 32));
  return v;
}
__device__ inline float rowsum16(float v) {
#pragma unroll
  for (int o = 1; o < 16; o <<= 1) v += __shfl_xor(v, o, 32);
  return v;
}

// ---------------- elementwise / prep kernels ----------------
__global__ void zero_counts_kernel(int* counts) {
  if (threadIdx.x < Ee) counts[threadIdx.x] = 0;
}

__global__ void cast_bf16_kernel(const float* __restrict__ in, bf16_t* __restrict__ out, int n) {
  int i = blockIdx.x * blockDim.x + threadIdx.x;
  if (i < n) out[i] = (bf16_t)in[i];
}

// in: [E][R][C] f32 -> out: [E][C][R] bf16
__global__ void cast_tr_kernel(const float* __restrict__ in, bf16_t* __restrict__ out,
                               int R, int C) {
  int e = blockIdx.z;
  int id = blockIdx.x * blockDim.x + threadIdx.x;
  if (id >= R * C) return;
  int r = id / C, c = id % C;
  out[((size_t)e * C + c) * R + r] = (bf16_t)in[(size_t)e * R * C + id];
}

__global__ __launch_bounds__(256) void rmsnorm_kernel(const float* __restrict__ in,
                                                      const float* __restrict__ w,
                                                      bf16_t* __restrict__ obf,
                                                      float* __restrict__ of) {
  int t = blockIdx.x;
  const float* x = in + (size_t)t * Hh;
  float ss = 0.f;
  for (int i = threadIdx.x; i < Hh; i += 256) { float v = x[i]; ss += v * v; }
  __shared__ float red[8];
#pragma unroll
  for (int o = 16; o > 0; o >>= 1) ss += __shfl_xor(ss, o, 32);
  if ((threadIdx.x & 31) == 0) red[threadIdx.x >> 5] = ss;
  __syncthreads();
  if (threadIdx.x == 0) {
    float tot = 0.f;
#pragma unroll
    for (int i = 0; i < 8; ++i) tot += red[i];
    red[0] = rsqrtf(tot / (float)Hh + EPSc);
  }
  __syncthreads();
  float rinv = red[0];
  for (int i = threadIdx.x; i < Hh; i += 256) {
    float v = x[i] * rinv * w[i];
    obf[(size_t)t * Hh + i] = (bf16_t)v;
    if (of) of[(size_t)t * Hh + i] = v;
  }
}

// qkv_f [T][1536] -> rope'd q_bf [B][NH][S][64], k_bf [B][NKV][S][64], vT_bf [B][NKV][64][S]
__global__ __launch_bounds__(256) void rope_pack_kernel(const float* __restrict__ qkvf,
                                                        const float* __restrict__ cosp,
                                                        const float* __restrict__ sinp,
                                                        bf16_t* __restrict__ qbf,
                                                        bf16_t* __restrict__ kbf,
                                                        bf16_t* __restrict__ vtbf) {
  int t = blockIdx.x;
  int b = t / Ss, sidx = t % Ss;
  const float* src = qkvf + (size_t)t * QKVN;
  const float* cr = cosp + (size_t)t * 32;
  const float* sr = sinp + (size_t)t * 32;
  for (int idx = threadIdx.x; idx < NHq * 32; idx += 256) {
    int hh = idx >> 5, d = idx & 31;
    float x1 = src[hh * 64 + d], x2 = src[hh * 64 + d + 32];
    float c = cr[d], sn = sr[d];
    size_t o = (((size_t)(b * NHq + hh)) * Ss + sidx) * HDd;
    qbf[o + d]      = (bf16_t)(x1 * c - x2 * sn);
    qbf[o + d + 32] = (bf16_t)(x2 * c + x1 * sn);
  }
  for (int idx = threadIdx.x; idx < NKVv * 32; idx += 256) {
    int hh = idx >> 5, d = idx & 31;
    float x1 = src[1024 + hh * 64 + d], x2 = src[1024 + hh * 64 + d + 32];
    float c = cr[d], sn = sr[d];
    size_t o = (((size_t)(b * NKVv + hh)) * Ss + sidx) * HDd;
    kbf[o + d]      = (bf16_t)(x1 * c - x2 * sn);
    kbf[o + d + 32] = (bf16_t)(x2 * c + x1 * sn);
  }
  for (int idx = threadIdx.x; idx < NKVv * HDd; idx += 256) {
    int hh = idx >> 6, d = idx & 63;
    vtbf[(((size_t)(b * NKVv + hh)) * HDd + d) * Ss + sidx] = (bf16_t)src[1280 + idx];
  }
}

// ---------------- WMMA GEMM kernels (one wave32 per 16x16 C tile) ----------------
__global__ __launch_bounds__(32) void gemm_qkv_kernel(const bf16_t* __restrict__ A,
                                                      const bf16_t* __restrict__ W,
                                                      float* __restrict__ C) {
  int lane = threadIdx.x, lr = lane & 15, grp = lane >> 4;
  int mt = blockIdx.x * 16, nt = blockIdx.y * 16;
  const bf16_t* ar = A + (size_t)(mt + lr) * Hh;
  const bf16_t* br = W + (size_t)(nt + lr) * Hh;
  v8f c = zero8();
  for (int kk = 0; kk < Hh; kk += 32)
    c = wmma_bf(load_a32(ar, kk, grp), load_b32(br, kk, grp), c);
#pragma unroll
  for (int r = 0; r < 8; ++r)
    C[(size_t)(mt + r + 8 * grp) * QKVN + nt + lr] = c[r];
}

__global__ __launch_bounds__(32) void gemm_oproj_kernel(const bf16_t* __restrict__ A,
                                                        const bf16_t* __restrict__ W,
                                                        const float* __restrict__ resid,
                                                        float* __restrict__ out) {
  int lane = threadIdx.x, lr = lane & 15, grp = lane >> 4;
  int mt = blockIdx.x * 16, nt = blockIdx.y * 16;
  const bf16_t* ar = A + (size_t)(mt + lr) * Hh;
  const bf16_t* br = W + (size_t)(nt + lr) * Hh;
  v8f c = zero8();
  for (int kk = 0; kk < Hh; kk += 32)
    c = wmma_bf(load_a32(ar, kk, grp), load_b32(br, kk, grp), c);
#pragma unroll
  for (int r = 0; r < 8; ++r) {
    size_t idx = (size_t)(mt + r + 8 * grp) * Hh + nt + lr;
    out[idx] = resid[idx] + c[r];
  }
}

// gathered MoE gate_up GEMM + clamp/GLU epilogue -> act_bf [E][T][I]
__global__ __launch_bounds__(32) void gemm_gateup_kernel(const bf16_t* __restrict__ X,
                                                         const bf16_t* __restrict__ Wgu,
                                                         const float* __restrict__ gub,
                                                         const int* __restrict__ counts,
                                                         const int* __restrict__ slots,
                                                         bf16_t* __restrict__ act) {
  int e = blockIdx.z;
  int cnt = counts[e];
  int mt = blockIdx.x * 16;
  if (mt >= cnt) return;                    // uniform exit (EXEC stays full)
  int nt = blockIdx.y * 16;
  int lane = threadIdx.x, lr = lane & 15, grp = lane >> 4;
  int row = mt + lr; if (row >= cnt) row = cnt - 1;
  int tok = slots[e * TDIM + row];
  const bf16_t* ar = X + (size_t)tok * Hh;
  const bf16_t* br = Wgu + ((size_t)e * (2 * Ii) + nt + lr) * Hh;
  v8f c = zero8();
  for (int kk = 0; kk < Hh; kk += 32)
    c = wmma_bf(load_a32(ar, kk, grp), load_b32(br, kk, grp), c);
  float bias = gub[e * (2 * Ii) + nt + lr];
#pragma unroll
  for (int r = 0; r < 8; ++r) {
    float v  = c[r] + bias;
    float pv = __shfl_xor(v, 1, 32);        // pair even(gate)/odd(up) columns
    if ((lr & 1) == 0) {
      float gate = fminf(v, LIMITc);
      float up   = fminf(fmaxf(pv, -LIMITc), LIMITc);
      float glu  = gate / (1.f + __expf(-ALPHAc * gate));
      float a    = (up + 1.f) * glu;
      int rg = mt + r + 8 * grp;
      if (rg < cnt)
        act[((size_t)e * TDIM + rg) * Ii + (nt >> 1) + (lr >> 1)] = (bf16_t)a;
    }
  }
}

// MoE down GEMM + bias, weighted atomic scatter into output
__global__ __launch_bounds__(32) void gemm_down_kernel(const bf16_t* __restrict__ act,
                                                       const bf16_t* __restrict__ Wdn,
                                                       const float* __restrict__ dnb,
                                                       const int* __restrict__ counts,
                                                       const int* __restrict__ slots,
                                                       const float* __restrict__ slotw,
                                                       float* __restrict__ out) {
  int e = blockIdx.z;
  int cnt = counts[e];
  int mt = blockIdx.x * 16;
  if (mt >= cnt) return;
  int nt = blockIdx.y * 16;
  int lane = threadIdx.x, lr = lane & 15, grp = lane >> 4;
  int row = mt + lr; if (row >= cnt) row = cnt - 1;
  const bf16_t* ar = act + ((size_t)e * TDIM + row) * Ii;
  const bf16_t* br = Wdn + ((size_t)e * Hh + nt + lr) * Ii;
  v8f c = zero8();
  for (int kk = 0; kk < Ii; kk += 32)
    c = wmma_bf(load_a32(ar, kk, grp), load_b32(br, kk, grp), c);
  float bias = dnb[e * Hh + nt + lr];
#pragma unroll
  for (int r = 0; r < 8; ++r) {
    int rg = mt + r + 8 * grp;
    if (rg < cnt) {
      int tok = slots[e * TDIM + rg];
      float wgt = slotw[e * TDIM + rg];
      atomicAdd(&out[(size_t)tok * Hh + nt + lr], wgt * (c[r] + bias));
    }
  }
}

// ---------------- flash attention with sink (one wave per 16-query tile) ----------------
__global__ __launch_bounds__(32) void attn_kernel(const bf16_t* __restrict__ qbf,
                                                  const bf16_t* __restrict__ kbf,
                                                  const bf16_t* __restrict__ vtbf,
                                                  const float* __restrict__ sinks,
                                                  bf16_t* __restrict__ attnbf) {
  __shared__ bf16_t Plds[16 * 32];
  const int lane = threadIdx.x, lr = lane & 15, grp = lane >> 4;
  const int qt = blockIdx.x * 16, h = blockIdx.y, b = blockIdx.z;
  const int kvh = h >> 2;
  const bf16_t* qbase = qbf  + ((size_t)(b * NHq  + h  )) * Ss * HDd;
  const bf16_t* kbase = kbf  + ((size_t)(b * NKVv + kvh)) * Ss * HDd;
  const bf16_t* vtb   = vtbf + ((size_t)(b * NKVv + kvh)) * HDd * Ss;

  const bf16_t* qrow = qbase + (size_t)(qt + lr) * HDd;
  bf16x16 a0 = load_a32(qrow, 0, grp);
  bf16x16 a1 = load_a32(qrow, 32, grp);

  v8f acc0 = zero8(), acc1 = zero8(), acc2 = zero8(), acc3 = zero8();
  float m[8], s[8];
#pragma unroll
  for (int r = 0; r < 8; ++r) { m[r] = -3.0e38f; s[r] = 0.f; }

  for (int kt = 0; kt <= qt + 15; kt += 32) {
    v8f c0 = zero8(), c1 = zero8();
    { const bf16_t* kr = kbase + (size_t)(kt + lr) * HDd;
      c0 = wmma_bf(a0, load_b32(kr, 0, grp), c0);
      c0 = wmma_bf(a1, load_b32(kr, 32, grp), c0); }
    { const bf16_t* kr = kbase + (size_t)(kt + 16 + lr) * HDd;
      c1 = wmma_bf(a0, load_b32(kr, 0, grp), c1);
      c1 = wmma_bf(a1, load_b32(kr, 32, grp), c1); }

#pragma unroll
    for (int r = 0; r < 8; ++r) {
      int qg = qt + r + 8 * grp;
      float s0 = c0[r] * SCALEc + ((kt + lr)      > qg ? NEGc : 0.f);
      float s1 = c1[r] * SCALEc + ((kt + 16 + lr) > qg ? NEGc : 0.f);
      float mx = rowmax16(fmaxf(s0, s1));
      float mnew = fmaxf(m[r], mx);
      float p0 = __expf(s0 - mnew);
      float p1 = __expf(s1 - mnew);
      float ps = rowsum16(p0 + p1);
      float sc = __expf(m[r] - mnew);
      s[r] = s[r] * sc + ps;
      m[r] = mnew;
      acc0[r] *= sc; acc1[r] *= sc; acc2[r] *= sc; acc3[r] *= sc;
      Plds[(r + 8 * grp) * 32 + lr]      = (bf16_t)p0;
      Plds[(r + 8 * grp) * 32 + 16 + lr] = (bf16_t)p1;
    }
    __syncthreads();
    B16U up;
    up.h[0] = *reinterpret_cast<const bf16x8*>(&Plds[lr * 32 + 8 * grp]);
    up.h[1] = *reinterpret_cast<const bf16x8*>(&Plds[lr * 32 + 16 + 8 * grp]);
    bf16x16 ap = up.v;
    acc0 = wmma_bf(ap, load_b32(vtb + (size_t)( 0 + lr) * Ss, kt, grp), acc0);
    acc1 = wmma_bf(ap, load_b32(vtb + (size_t)(16 + lr) * Ss, kt, grp), acc1);
    acc2 = wmma_bf(ap, load_b32(vtb + (size_t)(32 + lr) * Ss, kt, grp), acc2);
    acc3 = wmma_bf(ap, load_b32(vtb + (size_t)(48 + lr) * Ss, kt, grp), acc3);
    __syncthreads();
  }

  float sink = sinks[h];
#pragma unroll
  for (int r = 0; r < 8; ++r) {
    float inv = 1.f / (s[r] + __expf(sink - m[r]));
    int qg = qt + r + 8 * grp;
    size_t orow = ((size_t)(b * Ss + qg)) * (NHq * HDd) + h * HDd;
    attnbf[orow +  0 + lr] = (bf16_t)(acc0[r] * inv);
    attnbf[orow + 16 + lr] = (bf16_t)(acc1[r] * inv);
    attnbf[orow + 32 + lr] = (bf16_t)(acc2[r] * inv);
    attnbf[orow + 48 + lr] = (bf16_t)(acc3[r] * inv);
  }
}

// ---------------- router (top-2 of 8, softmax weights) ----------------
__global__ __launch_bounds__(256) void router_kernel(const float* __restrict__ x2f,
                                                     const float* __restrict__ rw,
                                                     const float* __restrict__ rb,
                                                     int* __restrict__ counts,
                                                     int* __restrict__ slots,
                                                     float* __restrict__ slotw) {
  int t = blockIdx.x;
  int e = threadIdx.x >> 5, lane = threadIdx.x & 31;
  const float* x = x2f + (size_t)t * Hh;
  const float* wv = rw + (size_t)e * Hh;
  float acc = 0.f;
  for (int i = lane; i < Hh; i += 32) acc += x[i] * wv[i];
#pragma unroll
  for (int o = 16; o > 0; o >>= 1) acc += __shfl_xor(acc, o, 32);
  __shared__ float logits[Ee];
  if (lane == 0) logits[e] = acc + rb[e];
  __syncthreads();
  if (threadIdx.x == 0) {
    int i0 = 0; float v0 = logits[0];
    for (int i = 1; i < Ee; ++i) if (logits[i] > v0) { v0 = logits[i]; i0 = i; }
    int i1 = -1; float v1 = -3e38f;
    for (int i = 0; i < Ee; ++i) if (i != i0 && logits[i] > v1) { v1 = logits[i]; i1 = i; }
    float p0 = 1.f / (1.f + __expf(v1 - v0));
    float p1 = 1.f - p0;
    int s0 = atomicAdd(&counts[i0], 1);
    slots[i0 * TDIM + s0] = t; slotw[i0 * TDIM + s0] = p0;
    int s1 = atomicAdd(&counts[i1], 1);
    slots[i1 * TDIM + s1] = t; slotw[i1 * TDIM + s1] = p1;
  }
}

// ---------------- host orchestration ----------------
extern "C" void kernel_launch(void* const* d_in, const int* in_sizes, int n_in,
                              void* d_out, int out_size, void* d_ws, size_t ws_size,
                              hipStream_t stream) {
  (void)in_sizes; (void)n_in; (void)out_size; (void)ws_size;
  const float* hidden = (const float*)d_in[0];
  const float* cosp   = (const float*)d_in[1];
  const float* sinp   = (const float*)d_in[2];
  // d_in[3]: attention_mask (pure causal; reproduced analytically)
  const float* q_w  = (const float*)d_in[4];
  const float* k_w  = (const float*)d_in[5];
  const float* v_w  = (const float*)d_in[6];
  const float* o_w  = (const float*)d_in[7];
  const float* sinks = (const float*)d_in[8];
  const float* ln1  = (const float*)d_in[9];
  const float* ln2  = (const float*)d_in[10];
  const float* rw   = (const float*)d_in[11];
  const float* rb   = (const float*)d_in[12];
  const float* guw  = (const float*)d_in[13];
  const float* gub  = (const float*)d_in[14];
  const float* dw   = (const float*)d_in[15];
  const float* dnb  = (const float*)d_in[16];
  float* out = (float*)d_out;

  char* wsb = (char*)d_ws;
  size_t off = 0;
  auto take = [&](size_t bytes) -> char* {
    char* p = wsb + off;
    off = (off + bytes + 255) & ~(size_t)255;
    return p;
  };
  bf16_t* xbf    = (bf16_t*)take((size_t)TDIM * Hh * 2);
  bf16_t* wqkv   = (bf16_t*)take((size_t)QKVN * Hh * 2);
  float*  qkvf   = (float*) take((size_t)TDIM * QKVN * 4);
  bf16_t* qbf    = (bf16_t*)take((size_t)Bb * NHq  * Ss * HDd * 2);
  bf16_t* kbf    = (bf16_t*)take((size_t)Bb * NKVv * Ss * HDd * 2);
  bf16_t* vtbf   = (bf16_t*)take((size_t)Bb * NKVv * HDd * Ss * 2);
  bf16_t* attnbf = (bf16_t*)take((size_t)TDIM * Hh * 2);
  bf16_t* wo     = (bf16_t*)take((size_t)Hh * Hh * 2);
  bf16_t* x2bf   = (bf16_t*)take((size_t)TDIM * Hh * 2);
  float*  x2f    = (float*) take((size_t)TDIM * Hh * 4);
  bf16_t* wgu    = (bf16_t*)take((size_t)Ee * (2 * Ii) * Hh * 2);
  bf16_t* wdn    = (bf16_t*)take((size_t)Ee * Hh * Ii * 2);
  bf16_t* actbf  = (bf16_t*)take((size_t)Ee * TDIM * Ii * 2);
  int*    counts = (int*)   take(Ee * sizeof(int));
  int*    slots  = (int*)   take((size_t)Ee * TDIM * sizeof(int));
  float*  slotw  = (float*) take((size_t)Ee * TDIM * sizeof(float));

  // --- prep: zero counters, cast weights to bf16 (QKV packed; MoE transposed) ---
  zero_counts_kernel<<<1, 32, 0, stream>>>(counts);
  cast_bf16_kernel<<<(NHq*HDd*Hh + 255)/256, 256, 0, stream>>>(q_w, wqkv, NHq*HDd*Hh);
  cast_bf16_kernel<<<(NKVv*HDd*Hh + 255)/256, 256, 0, stream>>>(k_w, wqkv + (size_t)NHq*HDd*Hh, NKVv*HDd*Hh);
  cast_bf16_kernel<<<(NKVv*HDd*Hh + 255)/256, 256, 0, stream>>>(v_w, wqkv + (size_t)(NHq+NKVv)*HDd*Hh, NKVv*HDd*Hh);
  cast_bf16_kernel<<<(Hh*Hh + 255)/256, 256, 0, stream>>>(o_w, wo, Hh*Hh);
  cast_tr_kernel<<<dim3((Hh*2*Ii + 255)/256, 1, Ee), 256, 0, stream>>>(guw, wgu, Hh, 2*Ii);
  cast_tr_kernel<<<dim3((Ii*Hh + 255)/256, 1, Ee), 256, 0, stream>>>(dw, wdn, Ii, Hh);

  // --- attention block ---
  rmsnorm_kernel<<<TDIM, 256, 0, stream>>>(hidden, ln1, xbf, nullptr);
  gemm_qkv_kernel<<<dim3(TDIM/16, QKVN/16), 32, 0, stream>>>(xbf, wqkv, qkvf);
  rope_pack_kernel<<<TDIM, 256, 0, stream>>>(qkvf, cosp, sinp, qbf, kbf, vtbf);
  attn_kernel<<<dim3(Ss/16, NHq, Bb), 32, 0, stream>>>(qbf, kbf, vtbf, sinks, attnbf);
  gemm_oproj_kernel<<<dim3(TDIM/16, Hh/16), 32, 0, stream>>>(attnbf, wo, hidden, out);

  // --- MoE block (sparse: only the top-2 selected experts are computed) ---
  rmsnorm_kernel<<<TDIM, 256, 0, stream>>>(out, ln2, x2bf, x2f);
  router_kernel<<<TDIM, 256, 0, stream>>>(x2f, rw, rb, counts, slots, slotw);
  gemm_gateup_kernel<<<dim3(TDIM/16, (2*Ii)/16, Ee), 32, 0, stream>>>(x2bf, wgu, gub, counts, slots, actbf);
  gemm_down_kernel<<<dim3(TDIM/16, Hh/16, Ee), 32, 0, stream>>>(actbf, wdn, dnb, counts, slots, slotw, out);
}
